// STSGNBlock_89833535964020
// MI455X (gfx1250) — matrix operand order repaired and checked by reading, not compiled, not run on hardware
//
#include <hip/hip_runtime.h>

// ---------------- problem dims ----------------
#define NN 64
#define TT 64
#define MM 2
#define VV 25
#define C_IN 64
#define C_OUT 128
#define EMBH 64
#define NG (NN * TT * MM)     // 8192 graphs
#define ROWS (NN * MM * VV)   // 3200 tcn rows

// ---------------- workspace layout (bytes) ----------------
#define H_BYTES ((size_t)NN * MM * TT * VV * C_OUT * 2)  // 52,428,800 (f16 h)
#define OFF_H 0
#define OFF_EJW (H_BYTES)                       // 32*128 f32 (zero-padded rows)
#define OFF_EF (OFF_EJW + 32 * 128 * 4)         // 64*128 f32
#define OFF_WTOP (OFF_EF + 64 * 128 * 4)        // 64*128 f16 (ci-major)
#define OFF_WT (OFF_WTOP + 64 * 128 * 2)        // 3*128*128 f16 (k, ci, co)
#define OFF_SC (OFF_WT + 3 * 128 * 128 * 2)     // 128 f32
#define OFF_SH (OFF_SC + 128 * 4)               // 128 f32

typedef _Float16 h8 __attribute__((ext_vector_type(8)));
typedef _Float16 v16h __attribute__((ext_vector_type(16)));
typedef float v8f __attribute__((ext_vector_type(8)));

static __device__ __forceinline__ v8f wmma16(v16h a, v16h b, v8f c) {
  // D = A(16x32 f16) * B(32x16 f16) + C(16x16 f32)
  return __builtin_amdgcn_wmma_f32_16x16x32_f16(false, a, false, b, (short)0, c,
                                                false, false);
}

// A fragment (16M x 32K, f16). row = absolute M row for this lane (caller adds lane&15).
// element e: K = k0 + (e<8?0:16) + (lane>=16?8:0) + (e&7)  -> two 16B loads.
static __device__ __forceinline__ v16h frag_a(const _Float16* __restrict__ base,
                                              int stride, int row, int k0,
                                              int lhalf) {
  const _Float16* q = base + (size_t)row * stride + k0 + lhalf * 8;
  h8 lo = *(const h8*)q;
  h8 hi = *(const h8*)(q + 16);
  v16h r;
#pragma unroll
  for (int i = 0; i < 8; ++i) {
    r[i] = lo[i];
    r[i + 8] = hi[i];
  }
  return r;
}

// B fragment (32K x 16N, f16). N = n0 + (lane&15); element e: K = k0 + (e<8?0:16) + (lane>=16?8:0) + (e&7).
static __device__ __forceinline__ v16h frag_b(const _Float16* __restrict__ base,
                                              int stride, int k0, int n0,
                                              int lane) {
  const _Float16* q = base + (size_t)(k0 + (lane >> 4) * 8) * stride + n0 + (lane & 15);
  v16h r;
#pragma unroll
  for (int i = 0; i < 8; ++i) {
    r[i] = q[(size_t)i * stride];
    r[i + 8] = q[(size_t)(i + 16) * stride];
  }
  return r;
}

// ---------------- kernel 1: tiny precompute ----------------
__global__ __launch_bounds__(256) void stsgn_precompute(
    const float* __restrict__ ej_w1, const float* __restrict__ ej_b1,
    const float* __restrict__ ej_w2, const float* __restrict__ ej_b2,
    const float* __restrict__ ef_w1, const float* __restrict__ ef_b1,
    const float* __restrict__ ef_w2, const float* __restrict__ ef_b2,
    const float* __restrict__ gcn_w, const float* __restrict__ tcn_w,
    const float* __restrict__ tcn_b, const float* __restrict__ bn_gamma,
    const float* __restrict__ bn_beta, const float* __restrict__ bn_mean,
    const float* __restrict__ bn_var, float* __restrict__ ejW,
    float* __restrict__ ef, _Float16* __restrict__ Wtop,
    _Float16* __restrict__ Wt, float* __restrict__ scale,
    float* __restrict__ shift) {
  __shared__ float ejs[VV * C_IN];
  __shared__ float hf[TT * EMBH];
  const int tid = threadIdx.x;

  // ej = relu(relu(W1 + b1) @ W2 + b2)   (V x CIN)
  for (int i = tid; i < VV * C_IN; i += 256) {
    int v = i / C_IN, c = i % C_IN;
    float a = ej_b2[c];
    for (int h = 0; h < EMBH; ++h)
      a += fmaxf(ej_w1[v * EMBH + h] + ej_b1[h], 0.f) * ej_w2[h * C_IN + c];
    ejs[i] = fmaxf(a, 0.f);
  }
  // frame hidden = relu(ef_w1 + b1)      (T x EMBH)
  for (int i = tid; i < TT * EMBH; i += 256)
    hf[i] = fmaxf(ef_w1[i] + ef_b1[i % EMBH], 0.f);
  __syncthreads();

  // ejW = ej @ gcn_w[CIN:2*CIN, :]  (zero-padded to 32 rows)
  for (int i = tid; i < 32 * C_OUT; i += 256) {
    int v = i >> 7, co = i & 127;
    float a = 0.f;
    if (v < VV)
      for (int c = 0; c < C_IN; ++c)
        a += ejs[v * C_IN + c] * gcn_w[(C_IN + c) * C_OUT + co];
    ejW[i] = a;
  }
  // ef = relu(hf @ ef_w2 + b2)           (T x COUT)
  for (int i = tid; i < TT * C_OUT; i += 256) {
    int t = i >> 7, co = i & 127;
    float a = ef_b2[co];
    for (int h = 0; h < EMBH; ++h) a += hf[t * EMBH + h] * ef_w2[h * C_OUT + co];
    ef[i] = fmaxf(a, 0.f);
  }
  // W_top (top CIN rows of gcn_w) -> f16, ci-major (already ci*COUT+co)
  for (int i = tid; i < C_IN * C_OUT; i += 256) Wtop[i] = (_Float16)gcn_w[i];
  // tcn weights -> f16, [k][ci][co]
  for (int i = tid; i < 3 * C_OUT * C_OUT; i += 256) {
    int k = i / (C_OUT * C_OUT);
    int r = i - k * C_OUT * C_OUT;
    int ci = r >> 7, co = r & 127;
    Wt[i] = (_Float16)tcn_w[(co * C_OUT + ci) * 3 + k];
  }
  // BN folded with conv bias: y = relu(conv*scale + shift)
  for (int co = tid; co < C_OUT; co += 256) {
    float s = bn_gamma[co] * rsqrtf(bn_var[co] + 1e-5f);
    scale[co] = s;
    shift[co] = bn_beta[co] + (tcn_b[co] - bn_mean[co]) * s;
  }
}

// ---------------- kernel 2: fused GCN (one wave = one graph) ----------------
#define XS_STRIDE 72   // 64 + pad, 16B multiple, bank-rotating
#define GS_STRIDE 40   // 32 + pad
#define TS_STRIDE 136  // 128 + pad

__global__ __launch_bounds__(128) void stsgn_gcn(
    const float* __restrict__ x, const float* __restrict__ G,
    const float* __restrict__ ejW, const float* __restrict__ ef,
    const float* __restrict__ gcn_b, const _Float16* __restrict__ Wtop,
    _Float16* __restrict__ h_ws) {
  __shared__ _Float16 xs[4][32 * XS_STRIDE];
  __shared__ _Float16 gs[4][32 * GS_STRIDE];
  __shared__ _Float16 ts[4][32 * TS_STRIDE];

  const int wave = threadIdx.x >> 5;
  const int lane = threadIdx.x & 31;
  const int lrow = lane & 15;
  const int lhalf = lane >> 4;

  const int g = blockIdx.x * 4 + wave;  // graph id = (n*T + t)*M + m
  const int m = g % MM;
  const int t = (g / MM) % TT;
  const int n = g / (MM * TT);

  const float* xg = x + (size_t)g * (VV * C_IN);
  const float* Gg = G + (size_t)g * (VV * VV);
  _Float16* xsw = xs[wave];
  _Float16* gsw = gs[wave];
  _Float16* tsw = ts[wave];

  // stage x -> f16 LDS (rows 0..24), zero rows 25..31
  for (int i = lane; i < VV * 32; i += 32) {  // 800 pairs
    int v = i >> 5, p = (i & 31) * 2;
    xsw[v * XS_STRIDE + p] = (_Float16)xg[v * C_IN + p];
    xsw[v * XS_STRIDE + p + 1] = (_Float16)xg[v * C_IN + p + 1];
  }
  for (int i = lane; i < 7 * C_IN; i += 32) {
    int v = VV + (i >> 6), c = i & 63;
    xsw[v * XS_STRIDE + c] = (_Float16)0.f;
  }
  // stage G -> f16 LDS, zero-padded 32x32
  for (int i = lane; i < 32 * 32; i += 32)
    gsw[(i >> 5) * GS_STRIDE + (i & 31)] = (_Float16)0.f;
  for (int i = lane; i < VV * VV; i += 32)
    gsw[(i / VV) * GS_STRIDE + (i % VV)] = (_Float16)Gg[i];
  __syncthreads();

  // GEMM1: tmp(32x128) = xs(32x64) @ Wtop(64x128) + ejW
#pragma unroll
  for (int mt = 0; mt < 2; ++mt) {
    v16h a0 = frag_a(xsw, XS_STRIDE, mt * 16 + lrow, 0, lhalf);
    v16h a1 = frag_a(xsw, XS_STRIDE, mt * 16 + lrow, 32, lhalf);
#pragma unroll
    for (int ct = 0; ct < 8; ++ct) {
      v8f acc;
#pragma unroll
      for (int r = 0; r < 8; ++r)
        acc[r] = ejW[(mt * 16 + r + lhalf * 8) * C_OUT + ct * 16 + lrow];
      v16h b0 = frag_b(Wtop, C_OUT, 0, ct * 16, lane);
      acc = wmma16(a0, b0, acc);
      v16h b1 = frag_b(Wtop, C_OUT, 32, ct * 16, lane);
      acc = wmma16(a1, b1, acc);
#pragma unroll
      for (int r = 0; r < 8; ++r)
        tsw[(mt * 16 + r + lhalf * 8) * TS_STRIDE + ct * 16 + lrow] =
            (_Float16)acc[r];
    }
  }
  __syncthreads();

  // GEMM2: h = relu(G @ tmp + b) + ef[t] ; write f16 h (n,m,t,v,c)
  const size_t hbase = ((size_t)(n * MM + m) * TT + t) * (VV * C_OUT);
#pragma unroll
  for (int mt = 0; mt < 2; ++mt) {
    v16h ag = frag_a(gsw, GS_STRIDE, mt * 16 + lrow, 0, lhalf);
#pragma unroll
    for (int ct = 0; ct < 8; ++ct) {
      v16h b = frag_b(tsw, TS_STRIDE, 0, ct * 16, lane);
      v8f acc = {};
      acc = wmma16(ag, b, acc);
      const int co = ct * 16 + lrow;
      const float bb = gcn_b[co];
      const float ee = ef[t * C_OUT + co];
#pragma unroll
      for (int r = 0; r < 8; ++r) {
        int u = mt * 16 + r + lhalf * 8;
        if (u < VV) {
          float hv = fmaxf(acc[r] + bb, 0.f) + ee;
          h_ws[hbase + (size_t)u * C_OUT + co] = (_Float16)hv;
        }
      }
    }
  }
}

// ---------------- kernel 3: temporal conv as WMMA GEMMs ----------------
// Y^T(t,co) = sum_k H^T(t+k-1, ci) @ W_k^T(ci, co); BN+ReLU folded.
__global__ __launch_bounds__(256) void stsgn_tcn(
    const _Float16* __restrict__ h_ws, const _Float16* __restrict__ Wt,
    const float* __restrict__ scale, const float* __restrict__ shift,
    float* __restrict__ xout) {
  __shared__ float sc_s[C_OUT], sh_s[C_OUT];
  if (threadIdx.x < C_OUT) {
    sc_s[threadIdx.x] = scale[threadIdx.x];
    sh_s[threadIdx.x] = shift[threadIdx.x];
  }
  __syncthreads();

  const int wave = threadIdx.x >> 5;
  const int lane = threadIdx.x & 31;
  const int lrow = lane & 15;
  const int lhalf = lane >> 4;
  const int ttile = wave & 3;  // 16-wide t tile
  const int ch = wave >> 2;    // co half (0..63 / 64..127)

  const int row = blockIdx.x;  // (n*M + m)*V + v
  const int nm = row / VV;
  const int v = row - nm * VV;
  const _Float16* hrow =
      h_ws + (size_t)nm * (TT * VV * C_OUT) + (size_t)v * C_OUT;

  v8f acc[4] = {{}, {}, {}, {}};

#pragma unroll
  for (int k = 0; k < 3; ++k) {
    const int tA = ttile * 16 + lrow + k - 1;  // per-lane shifted t row
    const bool valid = (unsigned)tA < (unsigned)TT;
    const _Float16* wk = Wt + k * (C_OUT * C_OUT);
#pragma unroll
    for (int ks = 0; ks < 4; ++ks) {
      v16h a = {};
      if (valid) a = frag_a(hrow, VV * C_OUT, tA, ks * 32, lhalf);
#pragma unroll
      for (int cc = 0; cc < 4; ++cc) {
        v16h b = frag_b(wk, C_OUT, ks * 32, ch * 64 + cc * 16, lane);
        acc[cc] = wmma16(a, b, acc[cc]);
      }
    }
  }

  const int nI = nm >> 1, mI = nm & 1;
  float* obase = xout + (size_t)nI * (TT * MM * VV * C_OUT) +
                 (size_t)(mI * VV + v) * C_OUT;
#pragma unroll
  for (int cc = 0; cc < 4; ++cc) {
    const int co = ch * 64 + cc * 16 + lrow;
    const float s = sc_s[co], sh = sh_s[co];
#pragma unroll
    for (int r = 0; r < 8; ++r) {
      const int tO = ttile * 16 + r + lhalf * 8;
      obase[(size_t)tO * (MM * VV * C_OUT) + co] =
          fmaxf(acc[cc][r] * s + sh, 0.f);
    }
  }
}

// ---------------- kernel 4: adj temporal conv (small, f32 VALU) ----------------
__global__ __launch_bounds__(256) void stsgn_adj(const float* __restrict__ G,
                                                 const float* __restrict__ adj_w,
                                                 const float* __restrict__ adj_b,
                                                 float* __restrict__ gout) {
  __shared__ float w_s[VV * VV * 3];
  __shared__ float b_s[VV];
  for (int i = threadIdx.x; i < VV * VV * 3; i += 256) w_s[i] = adj_w[i];
  if (threadIdx.x < VV) b_s[threadIdx.x] = adj_b[threadIdx.x];
  __syncthreads();

  const int q = threadIdx.x >> 6;
  const int t = threadIdx.x & 63;
  const int row = blockIdx.x * 4 + q;  // (n*M + m)*V + u
  const int n = row / (MM * VV);
  const int rem = row - n * (MM * VV);
  const int m = rem / VV;
  const int u = rem - m * VV;

  float gv[3][VV];
#pragma unroll
  for (int k = 0; k < 3; ++k) {
    const int tk = t + k - 1;
    const bool ok = (unsigned)tk < (unsigned)TT;
    const float* gp =
        G + ((((size_t)n * TT + (ok ? tk : 0)) * MM + m) * VV + u) * VV;
#pragma unroll
    for (int vi = 0; vi < VV; ++vi) gv[k][vi] = ok ? gp[vi] : 0.f;
  }

  float* op = gout + ((((size_t)n * TT + t) * MM + m) * VV + u) * VV;
  for (int vo = 0; vo < VV; ++vo) {
    float a = b_s[vo];
    const float* wp = &w_s[vo * VV * 3];
#pragma unroll
    for (int vi = 0; vi < VV; ++vi)
      a += wp[vi * 3 + 0] * gv[0][vi] + wp[vi * 3 + 1] * gv[1][vi] +
           wp[vi * 3 + 2] * gv[2][vi];
    op[vo] = a;
  }
}

// ---------------- launch ----------------
extern "C" void kernel_launch(void* const* d_in, const int* in_sizes, int n_in,
                              void* d_out, int out_size, void* d_ws,
                              size_t ws_size, hipStream_t stream) {
  (void)in_sizes;
  (void)n_in;
  (void)out_size;
  (void)ws_size;
  const float* x = (const float*)d_in[0];
  const float* G = (const float*)d_in[1];
  const float* ej_w1 = (const float*)d_in[2];
  const float* ej_b1 = (const float*)d_in[3];
  const float* ej_w2 = (const float*)d_in[4];
  const float* ej_b2 = (const float*)d_in[5];
  const float* ef_w1 = (const float*)d_in[6];
  const float* ef_b1 = (const float*)d_in[7];
  const float* ef_w2 = (const float*)d_in[8];
  const float* ef_b2 = (const float*)d_in[9];
  const float* gcn_w = (const float*)d_in[10];
  const float* gcn_b = (const float*)d_in[11];
  const float* tcn_w = (const float*)d_in[12];
  const float* tcn_b = (const float*)d_in[13];
  const float* bn_gamma = (const float*)d_in[14];
  const float* bn_beta = (const float*)d_in[15];
  const float* bn_mean = (const float*)d_in[16];
  const float* bn_var = (const float*)d_in[17];
  const float* adj_w = (const float*)d_in[18];
  const float* adj_b = (const float*)d_in[19];

  char* ws = (char*)d_ws;
  _Float16* h_ws = (_Float16*)(ws + OFF_H);
  float* ejW = (float*)(ws + OFF_EJW);
  float* ef = (float*)(ws + OFF_EF);
  _Float16* Wtop = (_Float16*)(ws + OFF_WTOP);
  _Float16* Wt = (_Float16*)(ws + OFF_WT);
  float* scale = (float*)(ws + OFF_SC);
  float* shift = (float*)(ws + OFF_SH);

  float* xout = (float*)d_out;
  float* gout = xout + (size_t)NN * TT * MM * VV * C_OUT;

  stsgn_precompute<<<1, 256, 0, stream>>>(
      ej_w1, ej_b1, ej_w2, ej_b2, ef_w1, ef_b1, ef_w2, ef_b2, gcn_w, tcn_w,
      tcn_b, bn_gamma, bn_beta, bn_mean, bn_var, ejW, ef, Wtop, Wt, scale,
      shift);
  stsgn_gcn<<<NG / 4, 128, 0, stream>>>(x, G, ejW, ef, gcn_b, Wtop, h_ws);
  stsgn_tcn<<<ROWS, 256, 0, stream>>>(h_ws, Wt, scale, shift, xout);
  stsgn_adj<<<ROWS / 4, 256, 0, stream>>>(G, adj_w, adj_b, gout);
}
